// EdgeScore_35665408426739
// MI455X (gfx1250) — matrix operand order repaired and checked by reading, not compiled, not run on hardware
//
#include <hip/hip_runtime.h>
#include <hip/hip_bf16.h>
#include <math.h>

typedef __attribute__((ext_vector_type(16))) _Float16 v16h;
typedef __attribute__((ext_vector_type(8)))  float    v8f;

#define IN_CH  64
#define HID_CH 128
#define NSEG   4096

// ---------------------------------------------------------------------------
// Kernel 1: score[e] = relu(edge_attr[e,:] @ w1 + b1) @ w2 + b2
// One wave per 16-edge tile; f16 WMMA with f32 accumulation.
// ---------------------------------------------------------------------------
__global__ __launch_bounds__(256) void mlp_score_kernel(
    const float* __restrict__ edge_attr,
    const float* __restrict__ w1,   // [64,128] row-major
    const float* __restrict__ b1,   // [128]
    const float* __restrict__ w2,   // [128]
    const float* __restrict__ b2,   // [1]
    float* __restrict__ score,      // [E]
    int numTiles)
{
  const int lane = threadIdx.x & 31;
  const int wave = threadIdx.x >> 5;
  const int hi   = lane >> 4;       // half of the wave: 0 = lanes 0-15, 1 = lanes 16-31
  const int lm   = lane & 15;

  // ---- Build all B fragments (w1 -> f16) in registers; reused for every tile.
  // Fragment (nt,kt) is the 32x16 (KxN) block: N = nt*16 + lm,
// element i of the lane's v16h holds K = kt*32 + hi*8 + i (i<8), kt*32 + 16 + hi*8 + (i-8) (i>=8).
  v16h bfrag[16];
#pragma unroll
  for (int nt = 0; nt < 8; ++nt) {
#pragma unroll
    for (int kt = 0; kt < 2; ++kt) {
      v16h b;
      const int n     = nt * 16 + lm;
      const int kbase = kt * 32 + hi * 8;
#pragma unroll
      for (int i = 0; i < 16; ++i)
        b[i] = (_Float16)w1[(kbase + i + (i >= 8 ? 8 : 0)) * HID_CH + n];
      bfrag[nt * 2 + kt] = b;
    }
  }

  // Per-lane epilogue scalars for hidden column n = nt*16 + lm (hoisted).
  float b1v[8], w2v[8];
#pragma unroll
  for (int nt = 0; nt < 8; ++nt) {
    b1v[nt] = b1[nt * 16 + lm];
    w2v[nt] = w2[nt * 16 + lm];
  }
  const float b2s = b2[0];

  const int waveGlobal = blockIdx.x * 8 + wave;
  const int waveCount  = gridDim.x * 8;

  for (int t = waveGlobal; t < numTiles; t += waveCount) {
    const long rowBase = (long)t * 16;
    const long row     = rowBase + lm;          // this lane's edge row (per A layout)

    // ---- A fragments: 16 edges x 32 K each, packed per ISA 16-bit A layout.
    // lanes 0-15 : K = kt*32 + {0..7, 16..23};  lanes 16-31: K = kt*32 + {8..15, 24..31}
    v16h afrag[2];
#pragma unroll
    for (int kt = 0; kt < 2; ++kt) {
      const float4* p0 = (const float4*)(edge_attr + row * IN_CH + kt * 32 + hi * 8);
      const float4* p1 = (const float4*)(edge_attr + row * IN_CH + kt * 32 + 16 + hi * 8);
      float4 f0 = p0[0], f1 = p0[1];
      float4 g0 = p1[0], g1 = p1[1];
      v16h a;
      a[0]=(_Float16)f0.x;  a[1]=(_Float16)f0.y;  a[2]=(_Float16)f0.z;  a[3]=(_Float16)f0.w;
      a[4]=(_Float16)f1.x;  a[5]=(_Float16)f1.y;  a[6]=(_Float16)f1.z;  a[7]=(_Float16)f1.w;
      a[8]=(_Float16)g0.x;  a[9]=(_Float16)g0.y;  a[10]=(_Float16)g0.z; a[11]=(_Float16)g0.w;
      a[12]=(_Float16)g1.x; a[13]=(_Float16)g1.y; a[14]=(_Float16)g1.z; a[15]=(_Float16)g1.w;
      afrag[kt] = a;
    }

    // Prefetch next tile's A rows (global_prefetch_b8).
    {
      const long nrow = rowBase + (long)waveCount * 16 + lm;
      if (nrow < (long)numTiles * 16)
        __builtin_prefetch(edge_attr + nrow * IN_CH + hi * 32, 0, 0);
    }

    float sacc[8] = {0.f,0.f,0.f,0.f,0.f,0.f,0.f,0.f};
#pragma unroll
    for (int nt = 0; nt < 8; ++nt) {
      v8f c = {};
      c = __builtin_amdgcn_wmma_f32_16x16x32_f16(false, afrag[0], false, bfrag[nt*2+0],
                                                 (short)0, c, false, false);
      c = __builtin_amdgcn_wmma_f32_16x16x32_f16(false, afrag[1], false, bfrag[nt*2+1],
                                                 (short)0, c, false, false);
      // C layout: VGPR r, lane L -> h[M = r + hi*8][N = nt*16 + lm]
#pragma unroll
      for (int r = 0; r < 8; ++r) {
        float h = c[r] + b1v[nt];
        h = fmaxf(h, 0.0f);
        sacc[r] += h * w2v[nt];
      }
    }

    // Reduce each sacc[r] across the 16-lane half (sums over hidden columns).
#pragma unroll
    for (int r = 0; r < 8; ++r) {
#pragma unroll
      for (int m = 1; m < 16; m <<= 1)
        sacc[r] += __shfl_xor(sacc[r], m, 32);
    }

    // Lane hi*16 + j (j = 0..7) writes edge rowBase + hi*8 + j with sacc[j].
    float v = sacc[0];
#pragma unroll
    for (int r = 1; r < 8; ++r)
      v = ((lm & 7) == r) ? sacc[r] : v;
    if (lm < 8)
      score[rowBase + hi * 8 + lm] = v + b2s;
  }
}

// ---------------------------------------------------------------------------
// Kernel 2: per-segment softmax, in place on `score` (= d_out).
// batch is sorted; one block per segment, range found by binary search.
// ---------------------------------------------------------------------------
__global__ __launch_bounds__(256) void seg_softmax_kernel(
    const int* __restrict__ batch,
    float* __restrict__ score,
    int E)
{
  const int seg = blockIdx.x;

  int lo = 0, hb = E;
  while (lo < hb) { int mid = (lo + hb) >> 1; if (batch[mid] < seg) lo = mid + 1; else hb = mid; }
  const int start = lo;
  hb = E;
  while (lo < hb) { int mid = (lo + hb) >> 1; if (batch[mid] < seg + 1) lo = mid + 1; else hb = mid; }
  const int end = lo;

  __shared__ float redm[8];
  __shared__ float reds[8];
  __shared__ float bc[2];

  const int tid  = threadIdx.x;
  const int lane = tid & 31;
  const int wid  = tid >> 5;

  // Pass 1: segment max
  float m = -INFINITY;
  for (int i = start + tid; i < end; i += 256) m = fmaxf(m, score[i]);
#pragma unroll
  for (int x = 1; x < 32; x <<= 1) m = fmaxf(m, __shfl_xor(m, x, 32));
  if (lane == 0) redm[wid] = m;
  __syncthreads();
  if (tid == 0) {
    float mm = redm[0];
#pragma unroll
    for (int i = 1; i < 8; ++i) mm = fmaxf(mm, redm[i]);
    bc[0] = mm;
  }
  __syncthreads();
  m = bc[0];

  // Pass 2: sum of exp(score - max)
  float s = 0.0f;
  for (int i = start + tid; i < end; i += 256) s += expf(score[i] - m);
#pragma unroll
  for (int x = 1; x < 32; x <<= 1) s += __shfl_xor(s, x, 32);
  if (lane == 0) reds[wid] = s;
  __syncthreads();
  if (tid == 0) {
    float ss = 0.0f;
#pragma unroll
    for (int i = 0; i < 8; ++i) ss += reds[i];
    bc[1] = ss;
  }
  __syncthreads();
  const float den = bc[1] + 1e-16f;

  // Pass 3: normalize in place
  for (int i = start + tid; i < end; i += 256)
    score[i] = expf(score[i] - m) / den;
}

// ---------------------------------------------------------------------------
extern "C" void kernel_launch(void* const* d_in, const int* in_sizes, int n_in,
                              void* d_out, int out_size, void* d_ws, size_t ws_size,
                              hipStream_t stream) {
  const float* edge_attr = (const float*)d_in[0];
  const float* w1        = (const float*)d_in[1];
  const float* b1        = (const float*)d_in[2];
  const float* w2        = (const float*)d_in[3];
  const float* b2        = (const float*)d_in[4];
  const int*   batch     = (const int*)d_in[5];
  float*       out       = (float*)d_out;

  const int E        = in_sizes[0] / IN_CH;
  const int numTiles = E / 16;

  mlp_score_kernel<<<4096, 256, 0, stream>>>(edge_attr, w1, b1, w2, b2, out, numTiles);
  seg_softmax_kernel<<<NSEG, 256, 0, stream>>>(batch, out, E);
}
